// Attention_53343493816354
// MI455X (gfx1250) — matrix operand order repaired and checked by reading, not compiled, not run on hardware
//
#include <hip/hip_runtime.h>
#include <math.h>

typedef __bf16 bf16_t;
typedef __attribute__((ext_vector_type(16))) __bf16 v16bf;
typedef __attribute__((ext_vector_type(8)))  __bf16 v8bf;
typedef __attribute__((ext_vector_type(8)))  float  v8f;

// ---------------------------------------------------------------------------
// WMMA helper: D = A(16x32 bf16) * B(32x16 bf16) + C (f32)
// ---------------------------------------------------------------------------
__device__ __forceinline__ v8f wmma_bf16(v16bf a, v16bf b, v8f c) {
  return __builtin_amdgcn_wmma_f32_16x16x32_bf16(
      /*neg_a=*/false, a, /*neg_b=*/false, b,
      /*c_mod=*/(short)0, c, /*reuse_a=*/false, /*reuse_b=*/false);
}

// A-fragment loader: row-major source, lane = row (lane%16),
// K pattern: e -> koff+e (contiguous), koff = (lane<16 ? 0 : 16)
__device__ __forceinline__ v16bf load_a_frag(const bf16_t* rowk /* row ptr at k-chunk base */,
                                             int koff) {
  v8bf lo = *reinterpret_cast<const v8bf*>(rowk + koff);
  v8bf hi = *reinterpret_cast<const v8bf*>(rowk + koff + 8);
  v16bf a;
#pragma unroll
  for (int i = 0; i < 8; ++i) { a[i] = lo[i]; a[8 + i] = hi[i]; }
  return a;
}

// ---------------------------------------------------------------------------
// Generic WMMA GEMM: C[M,N] = A[M,K](bf16) * B[N,K](bf16)^T + bias
// One wave computes one 16x16 tile. M,N multiples of 16, K multiple of 32.
// ---------------------------------------------------------------------------
__global__ void gemm_bf16_wmma(const bf16_t* __restrict__ A,
                               const bf16_t* __restrict__ B,
                               float* __restrict__ C,
                               const float* __restrict__ bias,
                               int M, int N, int K) {
  const int wave = blockIdx.x * (blockDim.x >> 5) + (threadIdx.x >> 5);
  const int lane = threadIdx.x & 31;
  const int ntiles = N >> 4;
  const int mt = wave / ntiles;
  const int nt = wave - mt * ntiles;
  const int row  = lane & 15;
  const int half = lane >> 4;          // 0 or 1
  const int koff = half * 16;

  const bf16_t* arow = A + (size_t)(mt * 16 + row) * K;
  const bf16_t* brow = B + (size_t)(nt * 16 + row) * K + half * 16;

  v8f acc = {};
  for (int kk = 0; kk < K; kk += 32) {
    v16bf a = load_a_frag(arow + kk, koff);
    v16bf b = *reinterpret_cast<const v16bf*>(brow + kk);
    acc = wmma_bf16(a, b, acc);
  }

  const int col = nt * 16 + row;
  const float bv = bias ? bias[col] : 0.0f;
#pragma unroll
  for (int r = 0; r < 8; ++r) {
    const int m = mt * 16 + r + half * 8;
    C[(size_t)m * N + col] = acc[r] + bv;
  }
}

// ---------------------------------------------------------------------------
// Flash attention: one wave = one (head, 16-query tile). bf16 in, online
// softmax, f32 accumulators, bf16 out into the concat buffer.
//   Qb : [8192, 256]  (global-head channels)
//   Kb : [Mkeys, 128] (branch-local channels)
//   Vt : [128, Mkeys] (transposed, depthwise-conv already added)
//   Out: [8192, 256]  concat buffer, branch selects column half
// ---------------------------------------------------------------------------
__global__ void flash_attn(const bf16_t* __restrict__ Qb,
                           const bf16_t* __restrict__ Kb,
                           const bf16_t* __restrict__ Vt,
                           bf16_t* __restrict__ Out,
                           int Mkeys, int branch) {
  __shared__ __align__(32) bf16_t Pld[4][16 * 32];   // per-wave P tile

  const int widx = threadIdx.x >> 5;
  const int lane = threadIdx.x & 31;
  const int wave = blockIdx.x * 4 + widx;
  const int h    = wave & 3;           // branch-local head
  const int qt   = wave >> 2;          // query tile (0..511)
  const int row  = lane & 15;
  const int half = lane >> 4;
  const int koff = half * 16;
  const int chb  = h * 32;             // channel base within branch's 128

  // ---- Q A-fragment (persistent) ----
  const bf16_t* qrow = Qb + (size_t)(qt * 16 + row) * 256 + (branch * 4 + h) * 32;
  const v16bf aq = load_a_frag(qrow, koff);

  float mrow[8], lsum[8];
#pragma unroll
  for (int r = 0; r < 8; ++r) { mrow[r] = -1e30f; lsum[r] = 0.0f; }
  v8f o0 = {}, o1 = {};

  const float scale = 0.17677669529663687f;   // 1/sqrt(32)
  bf16_t* P = Pld[widx];

  for (int kb = 0; kb < Mkeys; kb += 32) {
    // ---- S = Q K^T (two 16-key tiles) ----
    const v16bf bk0 = *reinterpret_cast<const v16bf*>(
        Kb + (size_t)(kb + row) * 128 + chb + half * 16);
    const v16bf bk1 = *reinterpret_cast<const v16bf*>(
        Kb + (size_t)(kb + 16 + row) * 128 + chb + half * 16);
    v8f zero = {};
    v8f s0 = wmma_bf16(aq, bk0, zero);
    v8f s1 = wmma_bf16(aq, bk1, zero);

    // ---- online softmax (per row r; 16-lane group = 16 key columns) ----
#pragma unroll
    for (int r = 0; r < 8; ++r) {
      float x0 = s0[r] * scale, x1 = s1[r] * scale;
      float mx = fmaxf(x0, x1);
      mx = fmaxf(mx, __shfl_xor(mx, 1, 32));
      mx = fmaxf(mx, __shfl_xor(mx, 2, 32));
      mx = fmaxf(mx, __shfl_xor(mx, 4, 32));
      mx = fmaxf(mx, __shfl_xor(mx, 8, 32));
      const float newm = fmaxf(mrow[r], mx);
      const float corr = __expf(mrow[r] - newm);
      const float p0 = __expf(x0 - newm);
      const float p1 = __expf(x1 - newm);
      float rs = p0 + p1;
      rs += __shfl_xor(rs, 1, 32);
      rs += __shfl_xor(rs, 2, 32);
      rs += __shfl_xor(rs, 4, 32);
      rs += __shfl_xor(rs, 8, 32);
      lsum[r] = lsum[r] * corr + rs;
      mrow[r] = newm;
      o0[r] *= corr;
      o1[r] *= corr;
      const int m = r + half * 8;
      P[m * 32 + row]      = (bf16_t)p0;   // keys kb..kb+15
      P[m * 32 + 16 + row] = (bf16_t)p1;   // keys kb+16..kb+31
    }

    // ---- P A-fragment via LDS (compiler inserts s_wait_dscnt) ----
    const bf16_t* prow = P + row * 32;
    v8bf plo = *reinterpret_cast<const v8bf*>(prow + koff);
    v8bf phi = *reinterpret_cast<const v8bf*>(prow + koff + 8);
    v16bf ap;
#pragma unroll
    for (int i = 0; i < 8; ++i) { ap[i] = plo[i]; ap[8 + i] = phi[i]; }

    // ---- O += P V (two 16-hd-column tiles) ----
    const v16bf bv0 = *reinterpret_cast<const v16bf*>(
        Vt + (size_t)(chb + row) * Mkeys + kb + half * 16);
    const v16bf bv1 = *reinterpret_cast<const v16bf*>(
        Vt + (size_t)(chb + 16 + row) * Mkeys + kb + half * 16);
    o0 = wmma_bf16(ap, bv0, o0);
    o1 = wmma_bf16(ap, bv1, o1);
  }

  // ---- normalize and write concat buffer ----
  const int colb = branch * 128 + chb;
#pragma unroll
  for (int r = 0; r < 8; ++r) {
    const float inv = 1.0f / lsum[r];
    const int q = qt * 16 + r + half * 8;
    Out[(size_t)q * 256 + colb + row]      = (bf16_t)(o0[r] * inv);
    Out[(size_t)q * 256 + colb + 16 + row] = (bf16_t)(o1[r] * inv);
  }
}

// ---------------------------------------------------------------------------
// Elementwise / data-movement kernels
// ---------------------------------------------------------------------------
__global__ void f2bf(const float* __restrict__ in, bf16_t* __restrict__ out, int n) {
  int i = blockIdx.x * blockDim.x + threadIdx.x;
  if (i < n) out[i] = (bf16_t)in[i];
}

// im2col for sr1: stride-4 4x4x1 patches.  A1[512, 4096] bf16
__global__ void im2col_sr1(const float* __restrict__ x, bf16_t* __restrict__ A1) {
  int i = blockIdx.x * blockDim.x + threadIdx.x;
  if (i >= 512 * 4096) return;
  int m = i >> 12, k = i & 4095;
  int c = k >> 4, r = k & 15, kh = r >> 2, kw = r & 3;
  int z = m & 7, hw = m >> 3, wo = hw & 7, ho = hw >> 3;
  int hh = ho * 4 + kh, ww = wo * 4 + kw;
  int n = (hh * 32 + ww) * 8 + z;
  A1[i] = (bf16_t)x[(size_t)n * 256 + c];
}

// im2col for sr2: stride-2 2x2x1 patches.  A2[2048, 1024] bf16
__global__ void im2col_sr2(const float* __restrict__ x, bf16_t* __restrict__ A2) {
  int i = blockIdx.x * blockDim.x + threadIdx.x;
  if (i >= 2048 * 1024) return;
  int m = i >> 10, k = i & 1023;
  int c = k >> 2, r = k & 3, kh = r >> 1, kw = r & 1;
  int z = m & 7, hw = m >> 3, wo = hw & 15, ho = hw >> 4;
  int hh = ho * 2 + kh, ww = wo * 2 + kw;
  int n = (hh * 32 + ww) * 8 + z;
  A2[i] = (bf16_t)x[(size_t)n * 256 + c];
}

// LayerNorm(256) + exact GELU, one wave per row, f32 in -> bf16 out
__global__ void ln_gelu(const float* __restrict__ X, const float* __restrict__ g,
                        const float* __restrict__ b, bf16_t* __restrict__ Y) {
  const int rowI = blockIdx.x;
  const int lane = threadIdx.x;   // 32 threads
  const float* x = X + (size_t)rowI * 256;
  float vals[8], s = 0.0f;
#pragma unroll
  for (int j = 0; j < 8; ++j) { vals[j] = x[lane + 32 * j]; s += vals[j]; }
#pragma unroll
  for (int m = 1; m < 32; m <<= 1) s += __shfl_xor(s, m, 32);
  const float mean = s * (1.0f / 256.0f);
  float vs = 0.0f;
#pragma unroll
  for (int j = 0; j < 8; ++j) { float d = vals[j] - mean; vs += d * d; }
#pragma unroll
  for (int m = 1; m < 32; m <<= 1) vs += __shfl_xor(vs, m, 32);
  const float inv = rsqrtf(vs * (1.0f / 256.0f) + 1e-5f);
#pragma unroll
  for (int j = 0; j < 8; ++j) {
    const int c = lane + 32 * j;
    float t = (vals[j] - mean) * inv * g[c] + b[c];
    float ge = 0.5f * t * (1.0f + erff(t * 0.70710678118654752f));
    Y[(size_t)rowI * 256 + c] = (bf16_t)ge;
  }
}

// Split K half of the KV GEMM output into bf16 [M,128]
__global__ void kv_split_k(const float* __restrict__ kv, bf16_t* __restrict__ Kb, int M) {
  int i = blockIdx.x * blockDim.x + threadIdx.x;
  if (i >= M * 128) return;
  int m = i >> 7, c = i & 127;
  Kb[i] = (bf16_t)kv[(size_t)m * 256 + c];
}

// V half: add depthwise 3x3x1 conv (pad 1,1,0), store transposed bf16 [128, M]
__global__ void dw_conv_v(const float* __restrict__ kv, const float* __restrict__ w,
                          const float* __restrict__ bias, bf16_t* __restrict__ Vt,
                          int M, int Hs, int Ws) {
  int i = blockIdx.x * blockDim.x + threadIdx.x;
  if (i >= M * 128) return;
  int ch = i / M;
  int m  = i - ch * M;
  int z = m & 7, hw = m >> 3;
  int ws = hw % Ws, hs = hw / Ws;
  float acc = bias[ch];
#pragma unroll
  for (int dh = 0; dh < 3; ++dh)
#pragma unroll
    for (int dw = 0; dw < 3; ++dw) {
      int hh = hs + dh - 1, ww = ws + dw - 1;
      if (hh >= 0 && hh < Hs && ww >= 0 && ww < Ws) {
        int mm = ((hh * Ws + ww) << 3) + z;
        acc += kv[(size_t)mm * 256 + 128 + ch] * w[ch * 9 + dh * 3 + dw];
      }
    }
  Vt[(size_t)ch * M + m] = (bf16_t)(kv[(size_t)m * 256 + 128 + ch] + acc);
}

// ---------------------------------------------------------------------------
extern "C" void kernel_launch(void* const* d_in, const int* in_sizes, int n_in,
                              void* d_out, int out_size, void* d_ws, size_t ws_size,
                              hipStream_t stream) {
  (void)in_sizes; (void)n_in; (void)out_size; (void)ws_size;
  const float* x      = (const float*)d_in[0];
  const float* Wq     = (const float*)d_in[1];
  const float* sr1_w  = (const float*)d_in[2];
  const float* sr1_b  = (const float*)d_in[3];
  const float* ln1_w  = (const float*)d_in[4];
  const float* ln1_b  = (const float*)d_in[5];
  const float* sr2_w  = (const float*)d_in[6];
  const float* sr2_b  = (const float*)d_in[7];
  const float* ln2_w  = (const float*)d_in[8];
  const float* ln2_b  = (const float*)d_in[9];
  const float* Wkv1   = (const float*)d_in[10];
  const float* Wkv2   = (const float*)d_in[11];
  const float* lc1_w  = (const float*)d_in[12];
  const float* lc1_b  = (const float*)d_in[13];
  const float* lc2_w  = (const float*)d_in[14];
  const float* lc2_b  = (const float*)d_in[15];
  const float* proj_w = (const float*)d_in[16];
  const float* proj_b = (const float*)d_in[17];
  float* out = (float*)d_out;

  // ---- workspace carve-up (256B aligned) ----
  char* ws = (char*)d_ws;
  size_t off = 0;
  auto alloc = [&](size_t bytes) -> char* {
    char* p = ws + off;
    off += (bytes + 255) & ~(size_t)255;
    return p;
  };
  bf16_t* xb    = (bf16_t*)alloc(8192 * 256 * 2);
  bf16_t* Wqb   = (bf16_t*)alloc(256 * 256 * 2);
  bf16_t* W1b   = (bf16_t*)alloc(256 * 4096 * 2);
  bf16_t* W2b   = (bf16_t*)alloc(256 * 1024 * 2);
  bf16_t* Wk1b  = (bf16_t*)alloc(256 * 256 * 2);
  bf16_t* Wk2b  = (bf16_t*)alloc(256 * 256 * 2);
  bf16_t* Wpb   = (bf16_t*)alloc(256 * 256 * 2);
  float*  Qf    = (float*) alloc((size_t)8192 * 256 * 4);
  bf16_t* Qbb   = (bf16_t*)alloc(8192 * 256 * 2);
  bf16_t* A1b   = (bf16_t*)alloc(512 * 4096 * 2);
  bf16_t* A2b   = (bf16_t*)alloc(2048 * 1024 * 2);
  float*  xr1f  = (float*) alloc(512 * 256 * 4);
  float*  xr2f  = (float*) alloc(2048 * 256 * 4);
  bf16_t* xr1b  = (bf16_t*)alloc(512 * 256 * 2);
  bf16_t* xr2b  = (bf16_t*)alloc(2048 * 256 * 2);
  float*  kv1f  = (float*) alloc(512 * 256 * 4);
  float*  kv2f  = (float*) alloc(2048 * 256 * 4);
  bf16_t* K1b   = (bf16_t*)alloc(512 * 128 * 2);
  bf16_t* K2b   = (bf16_t*)alloc(2048 * 128 * 2);
  bf16_t* Vt1b  = (bf16_t*)alloc(512 * 128 * 2);
  bf16_t* Vt2b  = (bf16_t*)alloc(2048 * 128 * 2);
  bf16_t* catb  = (bf16_t*)alloc(8192 * 256 * 2);

  auto cvt = [&](const float* src, bf16_t* dst, int n) {
    f2bf<<<(n + 255) / 256, 256, 0, stream>>>(src, dst, n);
  };
  auto gemm = [&](const bf16_t* A, const bf16_t* B, float* C, const float* bias,
                  int M, int N, int K) {
    int tiles = (M / 16) * (N / 16);
    gemm_bf16_wmma<<<tiles / 4, 128, 0, stream>>>(A, B, C, bias, M, N, K);
  };

  // 1) bf16 conversions (activations + weights)
  cvt(x,      xb,   8192 * 256);
  cvt(Wq,     Wqb,  256 * 256);
  cvt(sr1_w,  W1b,  256 * 4096);   // [O,I,4,4,1] is contiguous [256,4096]
  cvt(sr2_w,  W2b,  256 * 1024);
  cvt(Wkv1,   Wk1b, 256 * 256);
  cvt(Wkv2,   Wk2b, 256 * 256);
  cvt(proj_w, Wpb,  256 * 256);

  // 2) Q = x @ Wq^T   -> bf16
  gemm(xb, Wqb, Qf, nullptr, 8192, 256, 256);
  cvt(Qf, Qbb, 8192 * 256);

  // 3) branch spatial reductions: im2col -> WMMA GEMM (+conv bias) -> LN+GELU
  im2col_sr1<<<(512 * 4096 + 255) / 256, 256, 0, stream>>>(x, A1b);
  gemm(A1b, W1b, xr1f, sr1_b, 512, 256, 4096);
  ln_gelu<<<512, 32, 0, stream>>>(xr1f, ln1_w, ln1_b, xr1b);

  im2col_sr2<<<(2048 * 1024 + 255) / 256, 256, 0, stream>>>(x, A2b);
  gemm(A2b, W2b, xr2f, sr2_b, 2048, 256, 1024);
  ln_gelu<<<2048, 32, 0, stream>>>(xr2f, ln2_w, ln2_b, xr2b);

  // 4) KV projections; split K, depthwise-conv-add V and transpose
  gemm(xr1b, Wk1b, kv1f, nullptr, 512, 256, 256);
  kv_split_k<<<(512 * 128 + 255) / 256, 256, 0, stream>>>(kv1f, K1b, 512);
  dw_conv_v<<<(512 * 128 + 255) / 256, 256, 0, stream>>>(kv1f, lc1_w, lc1_b, Vt1b, 512, 8, 8);

  gemm(xr2b, Wk2b, kv2f, nullptr, 2048, 256, 256);
  kv_split_k<<<(2048 * 128 + 255) / 256, 256, 0, stream>>>(kv2f, K2b, 2048);
  dw_conv_v<<<(2048 * 128 + 255) / 256, 256, 0, stream>>>(kv2f, lc2_w, lc2_b, Vt2b, 2048, 16, 16);

  // 5) fused flash attention per branch (4 heads * 512 query tiles = 2048 waves)
  flash_attn<<<512, 128, 0, stream>>>(Qbb, K1b, Vt1b, catb, 512, 0);
  flash_attn<<<512, 128, 0, stream>>>(Qbb, K2b, Vt2b, catb, 2048, 1);

  // 6) output projection with bias -> d_out (f32)
  gemm(catb, Wpb, out, proj_b, 8192, 256, 256);
}
